// REGC_4398046511498
// MI455X (gfx1250) — matrix operand order repaired and checked by reading, not compiled, not run on hardware
//
#include <hip/hip_runtime.h>
#include <hip/hip_bf16.h>

// ---------------------------------------------------------------------------
// Types for CDNA5 WMMA (gfx1250, wave32): v_wmma_f32_16x16x32_bf16
// ---------------------------------------------------------------------------
typedef __attribute__((ext_vector_type(16))) __bf16 v16bf;
typedef __attribute__((ext_vector_type(8)))  float  v8f;

#define NEDGE  7
#define NEPR   400000       // edges per relation
#define NT     385000       // total nodes (author+fos+inst+paper)

// Async LDS staging path (CDNA5 GLOBAL_LOAD_ASYNC_TO_LDS, tracked by ASYNCcnt)
#if defined(__AMDGCN__) && __has_builtin(__builtin_amdgcn_global_load_async_to_lds_b128)
#define GEMM_ASYNC 1
#else
#define GEMM_ASYNC 0
#endif

#if GEMM_ASYNC
// builtin signature wants pointers to gcc-vector int4 in global / LDS spaces
typedef int v4i_vs __attribute__((vector_size(16)));
typedef __attribute__((address_space(1))) v4i_vs gvec4;
typedef __attribute__((address_space(3))) v4i_vs lvec4;
__device__ __forceinline__ gvec4* to_global(const void* p) {
    return (gvec4*)(unsigned long long)p;
}
// flat address of an LDS object: low 32 bits are the LDS offset (ISA 10.2)
__device__ __forceinline__ lvec4* to_lds(const void* p) {
    return (lvec4*)(unsigned int)(unsigned long long)p;
}
#endif

__device__ __forceinline__ void wait_async_lds() {
#if GEMM_ASYNC
#if __has_builtin(__builtin_amdgcn_s_wait_asynccnt)
    __builtin_amdgcn_s_wait_asynccnt(0);
#else
    asm volatile("s_wait_asynccnt 0" ::: "memory");
#endif
#endif
}

// ---------------------------------------------------------------------------
// float <-> order-preserving unsigned mapping for atomic float max
// ---------------------------------------------------------------------------
__device__ __forceinline__ unsigned f2mono(float f) {
    unsigned u = __float_as_uint(f);
    return (u & 0x80000000u) ? ~u : (u | 0x80000000u);
}
__device__ __forceinline__ float mono2f(unsigned u) {
    return __uint_as_float((u & 0x80000000u) ? (u & 0x7fffffffu) : ~u);
}

// ---------------------------------------------------------------------------
// fp32 -> bf16 conversion kernels
// ---------------------------------------------------------------------------
__global__ void cvt_bf16_kernel(const float* __restrict__ in, __bf16* __restrict__ out, long n) {
    long i = (long)blockIdx.x * blockDim.x + threadIdx.x;
    if (i < n) out[i] = (__bf16)in[i];
}

// out[c*R + r] = in[r*C + c]   (for bases_w1 which multiplies as x @ W)
__global__ void cvt_bf16_T_kernel(const float* __restrict__ in, __bf16* __restrict__ out, int R, int C) {
    int i = blockIdx.x * blockDim.x + threadIdx.x;
    if (i < R * C) {
        int r = i / C, c = i % C;
        out[(long)c * R + r] = (__bf16)in[i];
    }
}

// ---------------------------------------------------------------------------
// bf16 WMMA GEMM: C[M,N] = A[M,K](bf16) * W[N,K](bf16)^T (+bias)(+=C)
// 256 threads = 8 waves; block tile 128(M) x 64(N); wave tile 32x32 =
// 4 x v_wmma_f32_16x16x32_bf16 per K-step with A/B register reuse.
// K staged through double-buffered LDS (async-to-LDS when available),
// one barrier per K-step.
// ---------------------------------------------------------------------------
#define BM 128
#define BN 64
#define BK 32

__global__ __launch_bounds__(256)
void gemm_bf16_kernel(const __bf16* __restrict__ A, const __bf16* __restrict__ W,
                      float* __restrict__ C, const float* __restrict__ bias,
                      int M, int N, int K, int accum)
{
    __shared__ __bf16 sA[2][BM * BK];
    __shared__ __bf16 sB[2][BN * BK];

    const int tid  = threadIdx.x;
    const int lane = tid & 31;
    const int wave = tid >> 5;
    const int mw   = wave >> 1;   // 0..3 : 32-row slice
    const int nw   = wave & 1;    // 0..1 : 32-col slice
    const int m0   = blockIdx.x * BM;
    const int n0   = blockIdx.y * BN;

    // staging assignment: A tile 128x32 by all 256 threads (16 bf16 each),
    // B tile 64x32 by threads 0..127.
    const int  arow  = tid >> 1;
    const int  ahalf = (tid & 1) << 4;           // element offset 0 / 16
    const int  gm    = m0 + arow;
    const bool aok   = gm < M;
    const int  brow  = (tid & 127) >> 1;
    const int  bhalf = (tid & 1) << 4;
    const int  gn    = n0 + brow;
    const bool bok   = (tid < 128) && (gn < N);

    const __bf16* gA = A + (long)(aok ? gm : 0) * K + ahalf;
    const __bf16* gW = W + (long)(bok ? gn : 0) * K + bhalf;

    // zero OOB staging slots once (row validity is K-invariant)
    if (!aok) {
        uint4 z = {0, 0, 0, 0};
#pragma unroll
        for (int b = 0; b < 2; ++b) {
            uint4* dp = (uint4*)(sA[b] + arow * BK + ahalf);
            dp[0] = z; dp[1] = z;
        }
    }
    if ((tid < 128) && !bok) {
        uint4 z = {0, 0, 0, 0};
#pragma unroll
        for (int b = 0; b < 2; ++b) {
            uint4* dp = (uint4*)(sB[b] + brow * BK + bhalf);
            dp[0] = z; dp[1] = z;
        }
    }

    auto stage = [&](int k0, int buf) {
#if GEMM_ASYNC
        if (aok) {
            gvec4* g = to_global(gA + k0);
            lvec4* l = to_lds(sA[buf] + arow * BK + ahalf);
            __builtin_amdgcn_global_load_async_to_lds_b128(g, l, 0, 0);
            __builtin_amdgcn_global_load_async_to_lds_b128(g, l, 16, 0);
        }
        if (bok) {
            gvec4* g = to_global(gW + k0);
            lvec4* l = to_lds(sB[buf] + brow * BK + bhalf);
            __builtin_amdgcn_global_load_async_to_lds_b128(g, l, 0, 0);
            __builtin_amdgcn_global_load_async_to_lds_b128(g, l, 16, 0);
        }
#else
        if (aok) {
            const uint4* sp = (const uint4*)(gA + k0);
            uint4 v0 = sp[0], v1 = sp[1];
            uint4* dp = (uint4*)(sA[buf] + arow * BK + ahalf);
            dp[0] = v0; dp[1] = v1;
        }
        if (bok) {
            const uint4* sp = (const uint4*)(gW + k0);
            uint4 v0 = sp[0], v1 = sp[1];
            uint4* dp = (uint4*)(sB[buf] + brow * BK + bhalf);
            dp[0] = v0; dp[1] = v1;
        }
#endif
    };

    // Fragment addressing per CDNA5 ISA 16-bit layouts:
    // A: lanes 0-15 hold K {0..7,16..23}, lanes 16-31 hold K {8..15,24..31}
    // B: same K order as A: lanes 0-15 hold K {0..7,16..23}, lanes 16-31 hold K {8..15,24..31}
    const int fm  = lane & 15;
    const int kbA = (lane < 16) ? 0 : 8;
    const int kbB = (lane < 16) ? 0 : 8;
    const int ar0 = (mw * 32 + fm) * BK;
    const int ar1 = (mw * 32 + 16 + fm) * BK;
    const int br0 = (nw * 32 + fm) * BK;
    const int br1 = (nw * 32 + 16 + fm) * BK;

    v8f acc00 = {}, acc01 = {}, acc10 = {}, acc11 = {};

    const int nsteps = K / BK;
    stage(0, 0);
    int buf = 0;
    for (int s = 0; s < nsteps; ++s) {
        wait_async_lds();       // this wave's async copies into buf complete
        __syncthreads();        // everyone's copies complete / prev reads done
        if (s + 1 < nsteps) stage((s + 1) * BK, buf ^ 1);

        union Frag { uint4 u[2]; v16bf v; } a0, a1, b0, b1;
        const __bf16* pa = sA[buf];
        const __bf16* pb = sB[buf];
        a0.u[0] = *(const uint4*)(pa + ar0 + kbA);
        a0.u[1] = *(const uint4*)(pa + ar0 + kbA + 16);
        a1.u[0] = *(const uint4*)(pa + ar1 + kbA);
        a1.u[1] = *(const uint4*)(pa + ar1 + kbA + 16);
        b0.u[0] = *(const uint4*)(pb + br0 + kbB);
        b0.u[1] = *(const uint4*)(pb + br0 + kbB + 16);
        b1.u[0] = *(const uint4*)(pb + br1 + kbB);
        b1.u[1] = *(const uint4*)(pb + br1 + kbB + 16);

        acc00 = __builtin_amdgcn_wmma_f32_16x16x32_bf16(false, a0.v, false, b0.v, (short)0, acc00, false, false);
        acc01 = __builtin_amdgcn_wmma_f32_16x16x32_bf16(false, a0.v, false, b1.v, (short)0, acc01, false, false);
        acc10 = __builtin_amdgcn_wmma_f32_16x16x32_bf16(false, a1.v, false, b0.v, (short)0, acc10, false, false);
        acc11 = __builtin_amdgcn_wmma_f32_16x16x32_bf16(false, a1.v, false, b1.v, (short)0, acc11, false, false);
        buf ^= 1;
    }

    // C/D layout: col = (lane&15); VGPR r -> row +r (lanes 0-15) or +8+r.
    auto store_tile = [&](v8f& acc, int im, int in_) {
        int col = n0 + nw * 32 + in_ * 16 + fm;
        if (col >= N) return;
        float bv = bias ? bias[col] : 0.0f;
        int rbase = m0 + mw * 32 + im * 16 + ((lane < 16) ? 0 : 8);
#pragma unroll
        for (int r = 0; r < 8; ++r) {
            int row = rbase + r;
            if (row < M) {
                long idx = (long)row * N + col;
                float v  = acc[r] + bv;
                if (accum) v += C[idx];
                C[idx] = v;
            }
        }
    };
    store_tile(acc00, 0, 0);
    store_tile(acc01, 0, 1);
    store_tile(acc10, 1, 0);
    store_tile(acc11, 1, 1);
}

// ---------------------------------------------------------------------------
// Layer-1 root einsum: out1[n, h*32+d] = sum_b w[n, h*4+b] * bases[n, b*32+d]
// ---------------------------------------------------------------------------
__global__ void l1_root_combine_kernel(const float* __restrict__ w,
                                       const float* __restrict__ bases,
                                       float* __restrict__ out1, int n)
{
    long i = (long)blockIdx.x * blockDim.x + threadIdx.x;
    if (i >= (long)n * 256) return;
    long node = i >> 8;
    int  c    = (int)(i & 255);
    int  h    = c >> 5;
    int  dd   = c & 31;
    const float* wp = w + node * 32 + h * 4;
    const float* bp = bases + node * 128 + dd;
    out1[i] = wp[0] * bp[0] + wp[1] * bp[32] + wp[2] * bp[64] + wp[3] * bp[96];
}

// ---------------------------------------------------------------------------
// Layer-1 edge scatter: one wave per edge; 32 lanes x float4 = 128 feats.
// mean via atomicAdd, max via monotonic-uint atomicMax.
// ---------------------------------------------------------------------------
__global__ void l1_scatter_kernel(const float* __restrict__ bases,
                                  const int* __restrict__ src, const int* __restrict__ dst,
                                  float* __restrict__ sum, unsigned* __restrict__ mmax,
                                  float* __restrict__ cnt, int nE)
{
    int e = blockIdx.x * (blockDim.x >> 5) + (threadIdx.x >> 5);
    if (e >= nE) return;
    int lane = threadIdx.x & 31;
    int s = src[e], d = dst[e];
    if (e + 8 < nE)  // look-ahead prefetch of a future source row
        __builtin_prefetch(bases + (long)src[e + 8] * 128 + lane * 4, 0, 0);
    float4 v = *(const float4*)(bases + (long)s * 128 + lane * 4);
    float* sp = sum + (long)d * 128 + lane * 4;
    atomicAdd(sp + 0, v.x); atomicAdd(sp + 1, v.y);
    atomicAdd(sp + 2, v.z); atomicAdd(sp + 3, v.w);
    unsigned* mp = mmax + (long)d * 128 + lane * 4;
    atomicMax(mp + 0, f2mono(v.x)); atomicMax(mp + 1, f2mono(v.y));
    atomicMax(mp + 2, f2mono(v.z)); atomicMax(mp + 3, f2mono(v.w));
    if (lane == 0) atomicAdd(cnt + d, 1.0f);
}

// ---------------------------------------------------------------------------
// Layer-1 edge einsum combine:
// agg[b<4] = mean base b, agg[b>=4] = max base b-4 (0 if no in-edges)
// out1[n, h*32+d] += sum_{b<8} w2[n, h*8+b] * agg_b[d]
// ---------------------------------------------------------------------------
__global__ void l1_edge_combine_kernel(const float* __restrict__ sum,
                                       const unsigned* __restrict__ mmax,
                                       const float* __restrict__ cnt,
                                       const float* __restrict__ w2,
                                       float* __restrict__ out1, int n)
{
    long i = (long)blockIdx.x * blockDim.x + threadIdx.x;
    if (i >= (long)n * 256) return;
    long node = i >> 8;
    int  c    = (int)(i & 255);
    int  h    = c >> 5;
    int  dd   = c & 31;
    float cn  = cnt[node];
    float inv = 1.0f / fmaxf(cn, 1.0f);
    bool  has = cn > 0.0f;
    const float*    sp = sum  + node * 128 + dd;
    const unsigned* mp = mmax + node * 128 + dd;
    const float*    wp = w2   + node * 64  + h * 8;
    float a = 0.0f;
#pragma unroll
    for (int b = 0; b < 4; ++b) a += wp[b] * (sp[b * 32] * inv);
#pragma unroll
    for (int b = 0; b < 4; ++b) a += wp[4 + b] * (has ? mono2f(mp[b * 32]) : 0.0f);
    out1[i] += a;
}

// ---------------------------------------------------------------------------
// ReLU + bf16 conversion (hidden activations feed layer-2 WMMA GEMMs)
// ---------------------------------------------------------------------------
__global__ void relu_cvt_kernel(const float* __restrict__ in, __bf16* __restrict__ out, long n) {
    long i = (long)blockIdx.x * blockDim.x + threadIdx.x;
    if (i < n) out[i] = (__bf16)fmaxf(in[i], 0.0f);
}

// ---------------------------------------------------------------------------
// Layer-2 edge scatter (mean only): gather bf16 row, accumulate fp32 sums.
// ---------------------------------------------------------------------------
__global__ void l2_scatter_kernel(const __bf16* __restrict__ x,
                                  const int* __restrict__ src, const int* __restrict__ dst,
                                  float* __restrict__ sum, float* __restrict__ cnt, int nE)
{
    int e = blockIdx.x * (blockDim.x >> 5) + (threadIdx.x >> 5);
    if (e >= nE) return;
    int lane = threadIdx.x & 31;
    int s = src[e], d = dst[e];
    if (e + 8 < nE)
        __builtin_prefetch(x + (long)src[e + 8] * 256 + lane * 8, 0, 0);
    const __bf16* xp = x + (long)s * 256 + lane * 8;
    float* sp = sum + (long)d * 256 + lane * 8;
#pragma unroll
    for (int j = 0; j < 8; ++j) atomicAdd(sp + j, (float)xp[j]);
    if (lane == 0) atomicAdd(cnt + d, 1.0f);
}

// mean = sum / max(cnt,1), converted to bf16 for the rel_w2 WMMA GEMM
__global__ void mean_cvt_kernel(const float* __restrict__ sum, const float* __restrict__ cnt,
                                __bf16* __restrict__ out, long n)
{
    long i = (long)blockIdx.x * blockDim.x + threadIdx.x;
    if (i < n) {
        long node = i >> 8;
        out[i] = (__bf16)(sum[i] / fmaxf(cnt[node], 1.0f));
    }
}

// ---------------------------------------------------------------------------
// Host-side orchestration
// ---------------------------------------------------------------------------
extern "C" void kernel_launch(void* const* d_in, const int* in_sizes, int n_in,
                              void* d_out, int out_size, void* d_ws, size_t ws_size,
                              hipStream_t stream)
{
    (void)in_sizes; (void)n_in; (void)out_size; (void)ws_size;

    const float* x_paper    = (const float*)d_in[0];
    const float* emb_author = (const float*)d_in[1];
    const float* emb_fos    = (const float*)d_in[2];
    const float* emb_inst   = (const float*)d_in[3];
    const float* bases_w1   = (const float*)d_in[4];
    const float* rel_w1     = (const float*)d_in[5];
    const float* rel_b1     = (const float*)d_in[6];
    const float* root_w1    = (const float*)d_in[7];
    const float* root_b1    = (const float*)d_in[8];
    const float* rel_w2     = (const float*)d_in[9];
    const float* root_w2    = (const float*)d_in[10];
    const float* root_b2    = (const float*)d_in[11];
    const int*   src_idx    = (const int*)d_in[12];
    const int*   dst_idx    = (const int*)d_in[13];
    float* out = (float*)d_out;

    static const int  NUMSv[4] = {200000, 30000, 5000, 150000}; // author,fos,inst,paper
    static const long off[4]   = {0, 200000, 230000, 235000};
    static const int  SRC_T[7] = {0, 2, 0, 3, 3, 3, 1};
    static const int  DST_T[7] = {2, 0, 3, 0, 3, 1, 3};

    // ---- carve workspace ----
    char* p = (char*)d_ws;
    auto carve = [&](size_t bytes) -> char* {
        char* r = p;
        p += (bytes + 255) & ~(size_t)255;
        return r;
    };
    __bf16*   xin    = (__bf16*)  carve((size_t)NT * 128 * 2);        // inputs bf16
    float*    basesb = (float*)   carve((size_t)NT * 128 * 4);        // bases (fp32)
    float*    w1r    = (float*)   carve((size_t)NT * 32 * 4);         // root_w1 proj
    float*    out1   = (float*)   carve((size_t)NT * 256 * 4);        // layer-1 accum
    __bf16*   xs2    = (__bf16*)  carve((size_t)NT * 256 * 2);        // relu hidden
    __bf16*   wbB    = (__bf16*)  carve((size_t)128 * 128 * 2);       // bases_w1^T
    __bf16*   wbR1   = (__bf16*)  carve((size_t)4 * 32 * 128 * 2);
    __bf16*   wbE1   = (__bf16*)  carve((size_t)7 * 64 * 128 * 2);
    __bf16*   wbR2   = (__bf16*)  carve((size_t)4 * 349 * 256 * 2);
    __bf16*   wbE2   = (__bf16*)  carve((size_t)7 * 349 * 256 * 2);
    float*    sum1   = (float*)   carve((size_t)200000 * 128 * 4);
    unsigned* mmax   = (unsigned*)carve((size_t)200000 * 128 * 4);
    float*    cnt    = (float*)   carve((size_t)200000 * 4);
    float*    w2buf  = (float*)   carve((size_t)200000 * 64 * 4);
    float*    sum2   = (float*)   carve((size_t)200000 * 256 * 4);
    __bf16*   meanb  = (__bf16*)  carve((size_t)200000 * 256 * 2);

    auto cvt = [&](const float* s, __bf16* d, long n) {
        cvt_bf16_kernel<<<dim3((unsigned)((n + 255) / 256)), 256, 0, stream>>>(s, d, n);
    };
    auto gemm = [&](const __bf16* A, const __bf16* Wm, float* Cm, const float* bias,
                    int M, int N, int K, int accum) {
        dim3 g((M + BM - 1) / BM, (N + BN - 1) / BN);
        gemm_bf16_kernel<<<g, 256, 0, stream>>>(A, Wm, Cm, bias, M, N, K, accum);
    };

    // ---- precision prep: fp32 -> bf16 (activations + all weights) ----
    cvt(emb_author, xin + off[0] * 128, (long)NUMSv[0] * 128);
    cvt(emb_fos,    xin + off[1] * 128, (long)NUMSv[1] * 128);
    cvt(emb_inst,   xin + off[2] * 128, (long)NUMSv[2] * 128);
    cvt(x_paper,    xin + off[3] * 128, (long)NUMSv[3] * 128);
    cvt_bf16_T_kernel<<<(128 * 128 + 255) / 256, 256, 0, stream>>>(bases_w1, wbB, 128, 128);
    cvt(root_w1, wbR1, 4L * 32 * 128);
    cvt(rel_w1,  wbE1, 7L * 64 * 128);
    cvt(root_w2, wbR2, 4L * 349 * 256);
    cvt(rel_w2,  wbE2, 7L * 349 * 256);

    // ---- layer 1: bases + root projection + root einsum ----
    gemm(xin, wbB, basesb, nullptr, NT, 128, 128, 0);
    for (int t = 0; t < 4; ++t)
        gemm(xin + off[t] * 128, wbR1 + (long)t * 32 * 128, w1r + off[t] * 32,
             root_b1 + t * 32, NUMSv[t], 32, 128, 0);
    {
        long n = (long)NT * 256;
        l1_root_combine_kernel<<<(unsigned)((n + 255) / 256), 256, 0, stream>>>(w1r, basesb, out1, NT);
    }

    // ---- layer 1: per-relation mean/max aggregation + einsum ----
    for (int e = 0; e < NEDGE; ++e) {
        int s = SRC_T[e], d = DST_T[e];
        int nD = NUMSv[d];
        (void)hipMemsetAsync(sum1, 0, (size_t)nD * 128 * 4, stream);
        (void)hipMemsetAsync(mmax, 0, (size_t)nD * 128 * 4, stream);  // 0 == monotonic -inf
        (void)hipMemsetAsync(cnt,  0, (size_t)nD * 4, stream);
        l1_scatter_kernel<<<(NEPR + 7) / 8, 256, 0, stream>>>(
            basesb + off[s] * 128, src_idx + (long)e * NEPR, dst_idx + (long)e * NEPR,
            sum1, mmax, cnt, NEPR);
        gemm(xin + off[d] * 128, wbE1 + (long)e * 64 * 128, w2buf,
             rel_b1 + e * 64, nD, 64, 128, 0);
        long n = (long)nD * 256;
        l1_edge_combine_kernel<<<(unsigned)((n + 255) / 256), 256, 0, stream>>>(
            sum1, mmax, cnt, w2buf, out1 + off[d] * 256, nD);
    }

    // ---- ReLU -> bf16 hidden ----
    {
        long n = (long)NT * 256;
        relu_cvt_kernel<<<(unsigned)((n + 255) / 256), 256, 0, stream>>>(out1, xs2, n);
    }

    // ---- layer 2: root GEMMs write logits (covers every output row) ----
    for (int t = 0; t < 4; ++t)
        gemm(xs2 + off[t] * 256, wbR2 + (long)t * 349 * 256, out + off[t] * 349,
             root_b2 + t * 349, NUMSv[t], 349, 256, 0);

    // ---- layer 2: per-relation mean + rel_w2 GEMM accumulate ----
    for (int e = 0; e < NEDGE; ++e) {
        int s = SRC_T[e], d = DST_T[e];
        int nD = NUMSv[d];
        (void)hipMemsetAsync(sum2, 0, (size_t)nD * 256 * 4, stream);
        (void)hipMemsetAsync(cnt,  0, (size_t)nD * 4, stream);
        l2_scatter_kernel<<<(NEPR + 7) / 8, 256, 0, stream>>>(
            xs2 + off[s] * 256, src_idx + (long)e * NEPR, dst_idx + (long)e * NEPR,
            sum2, cnt, NEPR);
        long n = (long)nD * 256;
        mean_cvt_kernel<<<(unsigned)((n + 255) / 256), 256, 0, stream>>>(sum2, cnt, meanb, n);
        gemm(meanb, wbE2 + (long)e * 349 * 256, out + off[d] * 349, nullptr, nD, 349, 256, 1);
    }
}
